// LSTM_62783831933045
// MI455X (gfx1250) — matrix operand order repaired and checked by reading, not compiled, not run on hardware
//
#include <hip/hip_runtime.h>
#include <hip/hip_bf16.h>
#include <math.h>
#include <stdint.h>

typedef __bf16 bf16;
typedef bf16  v16bf __attribute__((ext_vector_type(16)));
typedef float v8f   __attribute__((ext_vector_type(8)));
typedef int   v4i   __attribute__((ext_vector_type(4)));

// Problem dims (static)
constexpr int D_IN   = 1024;
constexpr int D_CELL = 2048;
constexpr int T      = 2048;
constexpr int M      = 4 * D_CELL;   // 8192 gate rows
constexpr int KT_X   = D_IN / 32;    // 32 k-tiles of 32 for the pre-GEMM

#if __has_builtin(__builtin_amdgcn_global_load_async_to_lds_b128)
#define HAVE_ASYNC_LDS 1
#endif

// ---------------------------------------------------------------------------
// bf16 helpers (bf16 -> f32 is a 16-bit shift; avoids cvt instructions)
// ---------------------------------------------------------------------------
__device__ __forceinline__ float bflo(uint32_t w) {
    union { uint32_t u; float f; } v; v.u = w << 16; return v.f;
}
__device__ __forceinline__ float bfhi(uint32_t w) {
    union { uint32_t u; float f; } v; v.u = w & 0xFFFF0000u; return v.f;
}

// ---------------------------------------------------------------------------
// Pack W_x (first 1024 cols of each gate weight) into WMMA A-fragment order:
// dest elem index = ((mTile*KT_X + kTile)*32 + lane)*16 + e
// lane L (0..31): row = mTile*16 + (L&15)
// k (within 32-tile) = 8*(L>>4) + (e&7) + ((e>>3)<<4)   [ISA 16-bit A layout]
// ---------------------------------------------------------------------------
__global__ void pack_a(const float* __restrict__ Wf, const float* __restrict__ Wi,
                       const float* __restrict__ Wc, const float* __restrict__ Wo,
                       bf16* __restrict__ Apack) {
    int idx = blockIdx.x * blockDim.x + threadIdx.x;   // one bf16 element
    int e    = idx & 15;
    int L    = (idx >> 4) & 31;
    int tile = idx >> 9;
    int kt   = tile & (KT_X - 1);
    int mt   = tile >> 5;
    int row  = mt * 16 + (L & 15);
    int k    = kt * 32 + ((L >> 4) * 8) + (e & 7) + ((e >> 3) << 4);
    const float* W = (row < D_CELL)       ? Wf
                   : (row < 2 * D_CELL)   ? Wi
                   : (row < 3 * D_CELL)   ? Wc : Wo;
    int cell = row & (D_CELL - 1);
    Apack[idx] = (bf16)W[(size_t)cell * (D_IN + D_CELL) + k];
}

// ---------------------------------------------------------------------------
// Pack x [D_IN, T] into WMMA B-fragment order (32x16 bf16 tiles):
// lane L: col n = nTile*16 + (L&15); k = kTile*32 + (L>>4)*8 + (e&7) + ((e>>3)<<4)
// ---------------------------------------------------------------------------
__global__ void pack_b(const float* __restrict__ x, bf16* __restrict__ Bpack) {
    int idx  = blockIdx.x * blockDim.x + threadIdx.x;
    int e    = idx & 15;
    int L    = (idx >> 4) & 31;
    int tile = idx >> 9;
    int kt   = tile & (KT_X - 1);
    int nt   = tile >> 5;
    int n    = nt * 16 + (L & 15);
    int k    = kt * 32 + ((L >> 4) * 8) + (e & 7) + ((e >> 3) << 4);
    Bpack[idx] = (bf16)x[(size_t)k * T + n];
}

// ---------------------------------------------------------------------------
// Pack W_h (cols 1024..3071) row-major bf16: Wh[gate][cell][k]
// ---------------------------------------------------------------------------
__global__ void pack_wh(const float* __restrict__ Wf, const float* __restrict__ Wi,
                        const float* __restrict__ Wc, const float* __restrict__ Wo,
                        bf16* __restrict__ Wh) {
    int idx  = blockIdx.x * blockDim.x + threadIdx.x;   // 4*2048*2048 total
    int k    = idx & (D_CELL - 1);
    int cell = (idx >> 11) & (D_CELL - 1);
    int g    = idx >> 22;
    const float* W = (g == 0) ? Wf : (g == 1) ? Wi : (g == 2) ? Wc : Wo;
    Wh[idx] = (bf16)W[(size_t)cell * (D_IN + D_CELL) + D_IN + k];
}

// ---------------------------------------------------------------------------
// Zero the c / h ping-pong state
// ---------------------------------------------------------------------------
__global__ void zero_state(float* __restrict__ p, int n) {
    int i = blockIdx.x * blockDim.x + threadIdx.x;
    if (i < n) p[i] = 0.0f;
}

// ---------------------------------------------------------------------------
// Pre-GEMM: G[8192,2048] = W_x(bf16) @ x(bf16), f32 accumulate via WMMA.
// One wave computes a 64x64 output tile: 4 A-frags x 4 B-frags = 16 WMMAs/k.
// ---------------------------------------------------------------------------
__global__ void __launch_bounds__(256) pre_gemm(const bf16* __restrict__ Apack,
                                                const bf16* __restrict__ Bpack,
                                                float* __restrict__ G) {
    int wave = (blockIdx.x * blockDim.x + threadIdx.x) >> 5;  // 0..4095
    int lane = threadIdx.x & 31;
    int mBlk = wave >> 5;   // 0..127 (64 rows each)
    int nBlk = wave & 31;   // 0..31  (64 cols each)

    v8f acc[4][4];
#pragma unroll
    for (int i = 0; i < 4; ++i)
#pragma unroll
        for (int j = 0; j < 4; ++j)
            acc[i][j] = v8f{0.f,0.f,0.f,0.f,0.f,0.f,0.f,0.f};

    for (int kt = 0; kt < KT_X; ++kt) {
        v16bf a[4], b[4];
#pragma unroll
        for (int i = 0; i < 4; ++i) {
            int mt = mBlk * 4 + i;
            a[i] = *(const v16bf*)(Apack + ((size_t)((mt * KT_X + kt) * 32) + lane) * 16);
        }
#pragma unroll
        for (int j = 0; j < 4; ++j) {
            int nt = nBlk * 4 + j;
            b[j] = *(const v16bf*)(Bpack + ((size_t)((nt * KT_X + kt) * 32) + lane) * 16);
        }
        if (kt + 1 < KT_X) {  // hint next k-tile into near caches (global_prefetch_b8)
            __builtin_prefetch(Apack + ((size_t)((mBlk * 4 * KT_X + kt + 1) * 32) + lane) * 16, 0, 3);
            __builtin_prefetch(Bpack + ((size_t)((nBlk * 4 * KT_X + kt + 1) * 32) + lane) * 16, 0, 3);
        }
#pragma unroll
        for (int i = 0; i < 4; ++i)
#pragma unroll
            for (int j = 0; j < 4; ++j)
                acc[i][j] = __builtin_amdgcn_wmma_f32_16x16x32_bf16(
                    false, a[i], false, b[j], (short)0, acc[i][j], false, false);
    }

    // Epilogue: C layout -> lane L: n = L&15; VGPR r: m = r + 8*(L>>4)
    int rowHalf = (lane >> 4) * 8;
    int col0    = lane & 15;
#pragma unroll
    for (int i = 0; i < 4; ++i)
#pragma unroll
        for (int j = 0; j < 4; ++j)
#pragma unroll
            for (int r = 0; r < 8; ++r) {
                int row = mBlk * 64 + i * 16 + rowHalf + r;
                int col = nBlk * 64 + j * 16 + col0;
                G[(size_t)row * T + col] = acc[i][j][r];
            }
}

// ---------------------------------------------------------------------------
// One LSTM timestep. Grid = 256 blocks x 256 threads = 2048 waves;
// wave w computes cell w: 4 gate dot-products of length 2048 against h_prev
// staged in padded LDS (stride-9 words -> conflict-free on 64 banks).
// h_prev is staged via GLOBAL_LOAD_ASYNC_TO_LDS_B128 (ASYNCcnt); each 16B
// chunk stays contiguous under the +1/8 pad (chunks never straddle an
// 8-float boundary).
// ---------------------------------------------------------------------------
__global__ void __launch_bounds__(256) lstm_step(
    const bf16*  __restrict__ Wh,     // [4][D_CELL][D_CELL] bf16
    const float* __restrict__ G,      // [M][T] pre-gates
    const float* __restrict__ bfg, const float* __restrict__ big,
    const float* __restrict__ bcg, const float* __restrict__ bog,
    const float* __restrict__ hprev, float* __restrict__ hnext,
    float* __restrict__ c, float* __restrict__ out, int t) {

    __shared__ float sh[D_CELL + D_CELL / 8];   // padded: addr = k + (k>>3)

#if defined(HAVE_ASYNC_LDS)
    for (int chunk = threadIdx.x; chunk < D_CELL / 4; chunk += 256) {
        int k0 = chunk * 4;                     // 16B-aligned, pad-safe
        int pk = k0 + (k0 >> 3);
        __builtin_amdgcn_global_load_async_to_lds_b128(
            (__attribute__((address_space(1))) v4i*)(hprev + k0),
            (__attribute__((address_space(3))) v4i*)(sh + pk),
            /*offset=*/0, /*cpol=*/0);
    }
#if __has_builtin(__builtin_amdgcn_s_wait_asynccnt)
    __builtin_amdgcn_s_wait_asynccnt(0);
#else
    asm volatile("s_wait_asynccnt 0x0" ::: "memory");
#endif
    __syncthreads();
#else
    for (int k = threadIdx.x; k < D_CELL; k += 256)
        sh[k + (k >> 3)] = hprev[k];
    __syncthreads();
#endif

    int lane = threadIdx.x & 31;
    int wave = threadIdx.x >> 5;
    int cell = blockIdx.x * 8 + wave;

    float acc[4] = {0.f, 0.f, 0.f, 0.f};

    for (int it = 0; it < D_CELL / 256; ++it) {     // 8 chunks of 8 k per lane
        int kb = it * 256 + lane * 8;
        int sb = kb + (kb >> 3);
        float hv[8];
#pragma unroll
        for (int j = 0; j < 8; ++j) hv[j] = sh[sb + j];
#pragma unroll
        for (int g = 0; g < 4; ++g) {
            const uint4 wq = *(const uint4*)(Wh + (size_t)(g * D_CELL + cell) * D_CELL + kb);
            float s = acc[g];
            s = fmaf(bflo(wq.x), hv[0], s);
            s = fmaf(bfhi(wq.x), hv[1], s);
            s = fmaf(bflo(wq.y), hv[2], s);
            s = fmaf(bfhi(wq.y), hv[3], s);
            s = fmaf(bflo(wq.z), hv[4], s);
            s = fmaf(bfhi(wq.z), hv[5], s);
            s = fmaf(bflo(wq.w), hv[6], s);
            s = fmaf(bfhi(wq.w), hv[7], s);
            acc[g] = s;
        }
    }

    // wave32 reduction
#pragma unroll
    for (int off = 16; off > 0; off >>= 1) {
#pragma unroll
        for (int g = 0; g < 4; ++g)
            acc[g] += __shfl_xor(acc[g], off, 32);
    }

    if (lane == 0) {
        float gf = acc[0] + G[(size_t)(0 * D_CELL + cell) * T + t] + bfg[cell];
        float gi = acc[1] + G[(size_t)(1 * D_CELL + cell) * T + t] + big[cell];
        float gc = acc[2] + G[(size_t)(2 * D_CELL + cell) * T + t] + bcg[cell];
        float go = acc[3] + G[(size_t)(3 * D_CELL + cell) * T + t] + bog[cell];
        float fg = 1.0f / (1.0f + __expf(-gf));
        float ig = 1.0f / (1.0f + __expf(-gi));
        float cd = tanhf(gc);
        float og = 1.0f / (1.0f + __expf(-go));
        float cn = fmaf(fg, c[cell], ig * cd);
        c[cell] = cn;
        float h = og * tanhf(cn);
        hnext[cell] = h;
        out[(size_t)cell * T + t] = h;
    }
}

// ---------------------------------------------------------------------------
// Host launcher
// ---------------------------------------------------------------------------
extern "C" void kernel_launch(void* const* d_in, const int* in_sizes, int n_in,
                              void* d_out, int out_size, void* d_ws, size_t ws_size,
                              hipStream_t stream) {
    const float* x   = (const float*)d_in[0];
    const float* Wf  = (const float*)d_in[1];
    const float* bfg = (const float*)d_in[2];
    const float* Wi  = (const float*)d_in[3];
    const float* big = (const float*)d_in[4];
    const float* Wc  = (const float*)d_in[5];
    const float* bcg = (const float*)d_in[6];
    const float* Wo  = (const float*)d_in[7];
    const float* bog = (const float*)d_in[8];
    float* out = (float*)d_out;

    char* ws = (char*)d_ws;
    bf16*  Apack = (bf16*)(ws);                               // 16 MB
    bf16*  Bpack = (bf16*)(ws + ((size_t)16 << 20));          //  4 MB
    bf16*  Whp   = (bf16*)(ws + ((size_t)20 << 20));          // 32 MB
    float* G     = (float*)(ws + ((size_t)52 << 20));         // 64 MB
    float* cbuf  = (float*)(ws + ((size_t)116 << 20));        //  8 KB
    float* h0    = cbuf + D_CELL;
    float* h1    = h0 + D_CELL;

    // 1) pack weights / activations into WMMA layouts (bf16)
    pack_a <<<(M * D_IN)        / 256, 256, 0, stream>>>(Wf, Wi, Wc, Wo, Apack);
    pack_b <<<(D_IN * T)        / 256, 256, 0, stream>>>(x, Bpack);
    pack_wh<<<(4 * D_CELL * D_CELL) / 256, 256, 0, stream>>>(Wf, Wi, Wc, Wo, Whp);
    zero_state<<<(3 * D_CELL + 255) / 256, 256, 0, stream>>>(cbuf, 3 * D_CELL);

    // 2) time-parallel pre-gates G = W_x @ x  (WMMA bf16 -> f32)
    pre_gemm<<<512, 256, 0, stream>>>(Apack, Bpack, G);

    // 3) sequential recurrence, one launch per timestep (ping-pong h)
    for (int t = 0; t < T; ++t) {
        const float* hp = (t & 1) ? h1 : h0;
        float*       hn = (t & 1) ? h0 : h1;
        lstm_step<<<D_CELL / 8, 256, 0, stream>>>(Whp, G, bfg, big, bcg, bog,
                                                  hp, hn, cbuf, out, t);
    }
}
